// LSTMAcceptor_88330297409628
// MI455X (gfx1250) — compile-verified, matrix operand order checked by reading, not run on hardware
//
#include <hip/hip_runtime.h>
#include <hip/hip_bf16.h>
#include <stdint.h>

// Problem constants (from reference)
#define B_   256
#define T_   512
#define E_   256
#define H_   512
#define G4_  2048   // 4*H
#define FH_  512
#define OUT_ 2
#define ROWS_PER_WG 16
#define THREADS 256

typedef __attribute__((ext_vector_type(16))) __bf16 bf16x16;
typedef __attribute__((ext_vector_type(8)))  float  floatx8;

union FragU { bf16x16 v; uint4 q[2]; };

__device__ __forceinline__ uint32_t f2bf1(float f) {
  uint32_t u = __float_as_uint(f);
  return (u + 0x7FFFu + ((u >> 16) & 1u)) >> 16;   // round-to-nearest-even bf16
}
__device__ __forceinline__ uint32_t packbf2(float a, float b) {
  return f2bf1(a) | (f2bf1(b) << 16);
}
__device__ __forceinline__ float bf2f(uint32_t h) { return __uint_as_float(h << 16); }
__device__ __forceinline__ float sigmoidf_(float x) { return 1.0f / (1.0f + __expf(-x)); }

// ---------------------------------------------------------------------------
// Pack a row-major f32 weight (Krows x Ncols) into bf16 WMMA B-fragment tiles.
// Tile (nt,kt) covers N=[nt*16,nt*16+16), K=[kt*32,kt*32+32).
// Per tile: 256 dwords laid out so lane L reads dwords [L*8, L*8+8) contiguously:
//   nn = L&15 (column), half = L>>4
//   v<4  -> K = kt*32 + half*8 + 2v   (+pair)
//   v>=4 -> K = kt*32 + 16 + half*8 + 2(v-4)
// permuteGates: packed column np maps to source column (np&3)*(Ncols/4) + (np>>2)
// so the 4 LSTM gates of hidden unit j are adjacent in packed-N space.
// ---------------------------------------------------------------------------
__global__ void pack_weights(const float* __restrict__ src, uint32_t* __restrict__ dst,
                             int Krows, int Ncols, int permuteGates) {
  int tid = blockIdx.x * blockDim.x + threadIdx.x;
  int KT = Krows >> 5;
  int total = (Krows >> 5) * (Ncols >> 4) * 256;
  if (tid >= total) return;
  int v    = tid & 7;
  int lane = (tid >> 3) & 31;
  int tile = tid >> 8;
  int kt = tile % KT;
  int nt = tile / KT;
  int nn   = lane & 15;
  int half = lane >> 4;
  int kk = (v < 4) ? (half * 8 + 2 * v) : (16 + half * 8 + 2 * (v - 4));
  int k  = kt * 32 + kk;
  int np = nt * 16 + nn;
  int n  = permuteGates ? ((np & 3) * (Ncols >> 2) + (np >> 2)) : np;
  float f0 = src[(size_t)k * Ncols + n];
  float f1 = src[(size_t)(k + 1) * Ncols + n];
  dst[tid] = packbf2(f0, f1);
}

// LDS layout (bytes)
#define OFF_G    0        // 16 x 2048 f32 gate staging      (131072)
#define OFF_XE   131072   // 16 x 256  bf16 embedded inputs  (8192)
#define OFF_H    139264   // 16 x 512  bf16 hidden state     (16384)
#define OFF_C    155648   // 16 x 512  f32  cell state       (32768)
#define OFF_BIAS 188416   // 2048 f32 gate-permuted bias     (8192)
#define OFF_B1   196608   // 512 f32                          (2048)
#define OFF_LEN  198656   // 16 int                           (64)
#define SMEM_BYTES 198720

__global__ __launch_bounds__(THREADS) void lstm_main(
    const int* __restrict__ seq, const float* __restrict__ emb,
    const uint32_t* __restrict__ WiP, const uint32_t* __restrict__ WhP,
    const float* __restrict__ b_lstm, const uint32_t* __restrict__ W1P,
    const float* __restrict__ b1, const float* __restrict__ W2,
    const float* __restrict__ b2, float* __restrict__ out) {
  extern __shared__ char smem[];
  float*    sG    = (float*)(smem + OFF_G);
  uint16_t* sXe   = (uint16_t*)(smem + OFF_XE);
  uint16_t* sH    = (uint16_t*)(smem + OFF_H);
  float*    sC    = (float*)(smem + OFF_C);
  float*    sBias = (float*)(smem + OFF_BIAS);
  float*    sB1   = (float*)(smem + OFF_B1);
  int*      sLen  = (int*)(smem + OFF_LEN);

  const int tid  = threadIdx.x;
  const int b0   = blockIdx.x * ROWS_PER_WG;
  const int lane = tid & 31;
  const int wave = tid >> 5;
  const int mn   = lane & 15;   // M row (A frag) or N col (B/D frag)
  const int half = lane >> 4;

  // ---- one-time init ----
  for (int i = tid; i < G4_; i += THREADS)
    sBias[i] = b_lstm[(i & 3) * H_ + (i >> 2)];            // gate-interleave permute
  for (int i = tid; i < FH_; i += THREADS) sB1[i] = b1[i];
  for (int i = tid; i < ROWS_PER_WG * H_; i += THREADS) { sC[i] = 0.0f; sH[i] = 0; }
  if (tid < ROWS_PER_WG) sLen[tid] = 0;
  __syncthreads();
  {  // per-row sequence lengths (prefix of non-PAD tokens)
    int r = tid >> 4, seg = tid & 15;
    const int* sp = seq + (size_t)(b0 + r) * T_ + seg * 32;
    int c = 0;
#pragma unroll
    for (int p = 0; p < 32; ++p) c += (sp[p] != 0);
    atomicAdd(&sLen[r], c);
  }
  __syncthreads();

  // ---- recurrence over time ----
  for (int t = 0; t < T_; ++t) {
    // (1) gather embeddings for this step -> bf16 LDS
    {
      int m = tid >> 4, seg = tid & 15;
      int tok = seq[(size_t)(b0 + m) * T_ + t];
      const float4* ep = (const float4*)(emb + (size_t)tok * E_ + seg * 16);
      float4 a0 = ep[0], a1 = ep[1], a2 = ep[2], a3 = ep[3];
      uint4 o0, o1;
      o0.x = packbf2(a0.x, a0.y); o0.y = packbf2(a0.z, a0.w);
      o0.z = packbf2(a1.x, a1.y); o0.w = packbf2(a1.z, a1.w);
      o1.x = packbf2(a2.x, a2.y); o1.y = packbf2(a2.z, a2.w);
      o1.z = packbf2(a3.x, a3.y); o1.w = packbf2(a3.z, a3.w);
      uint4* dp = (uint4*)(sXe + m * E_ + seg * 16);
      dp[0] = o0; dp[1] = o1;
    }
    __syncthreads();

    // (2a) preload ALL A-fragments for this step into registers (once, not per N-tile)
    FragU aXe[E_ / 32];
    FragU aH[H_ / 32];
#pragma unroll
    for (int kt = 0; kt < (E_ / 32); ++kt) {
      aXe[kt].q[0] = *(const uint4*)(sXe + mn * E_ + kt * 32 + half * 8);
      aXe[kt].q[1] = *(const uint4*)(sXe + mn * E_ + kt * 32 + 16 + half * 8);
    }
#pragma unroll
    for (int kt = 0; kt < (H_ / 32); ++kt) {
      aH[kt].q[0] = *(const uint4*)(sH + mn * H_ + kt * 32 + half * 8);
      aH[kt].q[1] = *(const uint4*)(sH + mn * H_ + kt * 32 + 16 + half * 8);
    }
    __syncthreads();   // all sH/sXe reads done -> gate phase below may rewrite them

    // (2b) fused GEMM: G = Xe@Wi + H@Wh + b  (wave owns 16 N-tiles = 256 packed cols)
    // inner loop now touches only global B-fragments + WMMA
    for (int nti = 0; nti < 16; ++nti) {
      int nt = wave * 16 + nti;
      float bias = sBias[nt * 16 + mn];
      floatx8 acc = {bias, bias, bias, bias, bias, bias, bias, bias};
#pragma unroll
      for (int kt = 0; kt < (E_ / 32); ++kt) {      // input contribution
        FragU b;
        const uint4* wp = (const uint4*)(WiP + ((size_t)(nt * (E_ / 32) + kt)) * 256 + lane * 8);
        b.q[0] = wp[0]; b.q[1] = wp[1];
        acc = __builtin_amdgcn_wmma_f32_16x16x32_bf16(false, aXe[kt].v, false, b.v,
                                                      (short)0, acc, false, false);
      }
#pragma unroll
      for (int kt = 0; kt < (H_ / 32); ++kt) {      // recurrent contribution
        FragU b;
        const uint4* wp = (const uint4*)(WhP + ((size_t)(nt * (H_ / 32) + kt)) * 256 + lane * 8);
        b.q[0] = wp[0]; b.q[1] = wp[1];
        acc = __builtin_amdgcn_wmma_f32_16x16x32_bf16(false, aH[kt].v, false, b.v,
                                                      (short)0, acc, false, false);
      }
      int m0 = half * 8, n = nt * 16 + mn;
#pragma unroll
      for (int r = 0; r < 8; ++r) sG[(m0 + r) * G4_ + n] = acc[r];
    }

    // (3) gate fusion: wave owns hidden units j in [wave*64, wave*64+64), all 16 rows.
    // Reads only own-wave sG region and own sH/sC columns -> no barrier needed here.
    for (int it = 0; it < 32; ++it) {
      int idx = it * 32 + lane;
      int m = idx >> 6;
      int j = wave * 64 + (idx & 63);
      int gb = m * G4_ + 4 * j;                  // gate-interleaved columns
      float ig = sigmoidf_(sG[gb + 0]);
      float fg = sigmoidf_(sG[gb + 1]);
      float gg = tanhf(sG[gb + 2]);
      float og = sigmoidf_(sG[gb + 3]);
      float c_old = sC[m * H_ + j];
      float c_new = fg * c_old + ig * gg;
      float h_new = og * tanhf(c_new);
      bool  act   = (t < sLen[m]);
      float h_old = bf2f(sH[m * H_ + j]);
      float cw = act ? c_new : c_old;
      float hw = act ? h_new : h_old;
      sC[m * H_ + j] = cw;
      sH[m * H_ + j] = (uint16_t)f2bf1(hw);
    }
    __syncthreads();   // gate writes visible before next step's preload
  }

  // ---- head: hidden = tanh(h_last @ W1 + b1), via WMMA (32 N-tiles, wave owns 4) ----
  {
    FragU aH[H_ / 32];
#pragma unroll
    for (int kt = 0; kt < (H_ / 32); ++kt) {
      aH[kt].q[0] = *(const uint4*)(sH + mn * H_ + kt * 32 + half * 8);
      aH[kt].q[1] = *(const uint4*)(sH + mn * H_ + kt * 32 + 16 + half * 8);
    }
    for (int nti = 0; nti < 4; ++nti) {
      int nt = wave * 4 + nti;
      float bias = sB1[nt * 16 + mn];
      floatx8 acc = {bias, bias, bias, bias, bias, bias, bias, bias};
#pragma unroll
      for (int kt = 0; kt < (H_ / 32); ++kt) {
        FragU b;
        const uint4* wp = (const uint4*)(W1P + ((size_t)(nt * (H_ / 32) + kt)) * 256 + lane * 8);
        b.q[0] = wp[0]; b.q[1] = wp[1];
        acc = __builtin_amdgcn_wmma_f32_16x16x32_bf16(false, aH[kt].v, false, b.v,
                                                      (short)0, acc, false, false);
      }
      int m0 = half * 8, n = nt * 16 + mn;
#pragma unroll
      for (int r = 0; r < 8; ++r) sG[(m0 + r) * FH_ + n] = tanhf(acc[r]);  // reuse sG
    }
  }
  __syncthreads();

  // ---- logits (FH x 2) + softmax: one wave, one lane per (row, class) ----
  if (tid < 32) {
    int m = tid >> 1, o = tid & 1;
    float acc = b2[o];
    for (int k = 0; k < FH_; ++k) acc += sG[m * FH_ + k] * W2[k * OUT_ + o];
    float other = __shfl_xor(acc, 1, 32);
    float mx = fmaxf(acc, other);
    float e0 = __expf(acc - mx), e1 = __expf(other - mx);
    out[(size_t)(b0 + m) * OUT_ + o] = e0 / (e0 + e1);
  }
}

extern "C" void kernel_launch(void* const* d_in, const int* in_sizes, int n_in,
                              void* d_out, int out_size, void* d_ws, size_t ws_size,
                              hipStream_t stream) {
  (void)in_sizes; (void)n_in; (void)out_size; (void)ws_size;
  const int*   seq = (const int*)d_in[0];
  const float* emb = (const float*)d_in[1];
  const float* Wi  = (const float*)d_in[2];
  const float* Wh  = (const float*)d_in[3];
  const float* bl  = (const float*)d_in[4];
  const float* W1  = (const float*)d_in[5];
  const float* b1  = (const float*)d_in[6];
  const float* W2  = (const float*)d_in[7];
  const float* b2  = (const float*)d_in[8];
  float* out = (float*)d_out;

  uint32_t* ws  = (uint32_t*)d_ws;
  uint32_t* WiP = ws;                         // 256*2048/2 = 262144 dwords (1 MB)
  uint32_t* WhP = ws + 262144;                // 512*2048/2 = 524288 dwords (2 MB)
  uint32_t* W1P = ws + 262144 + 524288;       // 512*512/2  = 131072 dwords (0.5 MB)

  pack_weights<<<262144 / 256, 256, 0, stream>>>(Wi, WiP, E_, G4_, 1);
  pack_weights<<<524288 / 256, 256, 0, stream>>>(Wh, WhP, H_, G4_, 1);
  pack_weights<<<131072 / 256, 256, 0, stream>>>(W1, W1P, H_, FH_, 0);

  lstm_main<<<B_ / ROWS_PER_WG, THREADS, SMEM_BYTES, stream>>>(
      seq, emb, WiP, WhP, bl, W1P, b1, W2, b2, out);
}
